// Attention_42975442764025
// MI455X (gfx1250) — matrix operand order repaired and checked by reading, not compiled, not run on hardware
//
#include <hip/hip_runtime.h>

// ---------- types ----------
typedef _Float16 h16;
typedef __attribute__((ext_vector_type(16))) _Float16 v16h;
typedef __attribute__((ext_vector_type(8)))  _Float16 v8h;
typedef __attribute__((ext_vector_type(4)))  _Float16 v4h;
typedef __attribute__((ext_vector_type(8)))  float    v8f;

#define NB 8
#define T  2048
#define D  1024

// CDNA5 async global->LDS copy (16B per lane), tracked by ASYNCcnt.
#define ASYNC_G2L_B128(ldsoff, gptr)                                  \
  asm volatile("global_load_async_to_lds_b128 %0, %1, off"            \
               :: "v"(ldsoff), "v"(gptr) : "memory")
// async loads complete in order: waiting <=5 retires the older phase's 5 copies
#define WAIT_ASYNC0() asm volatile("s_wait_asynccnt 0x0" ::: "memory")
#define WAIT_ASYNC5() asm volatile("s_wait_asynccnt 0x5" ::: "memory")

// ---------- WMMA helpers ----------
__device__ __forceinline__ v8f wmma_f16(v16h a, v16h b, v8f c) {
  return __builtin_amdgcn_wmma_f32_16x16x32_f16(
      /*neg_a=*/false, a, /*neg_b=*/false, b,
      /*c_mod=*/(short)0, c, /*reuse_a=*/false, /*reuse_b=*/false);
}

// A-matrix fragment (16x32 f16), storage [row][k] k-contiguous, pitch in halves.
__device__ __forceinline__ v16h load_frag_a(const h16* __restrict__ base, int pitch,
                                            int row, int kbase, int lane) {
  const h16* p = base + (size_t)(row + (lane & 15)) * pitch + kbase + ((lane >> 4) << 3);
  v8h lo = *(const v8h*)p;
  v8h hi = *(const v8h*)(p + 16);
  return __builtin_shufflevector(lo, hi, 0,1,2,3,4,5,6,7,8,9,10,11,12,13,14,15);
}

// B-matrix fragment (32x16 f16), storage [n][k] k-contiguous.
__device__ __forceinline__ v16h load_frag_b(const h16* __restrict__ base, int pitch,
                                            int col, int kbase, int lane) {
  const h16* p = base + (size_t)(col + (lane & 15)) * pitch + kbase + ((lane >> 4) << 3);
  v8h lo = *(const v8h*)p;
  v8h hi = *(const v8h*)(p + 16);
  return __builtin_shufflevector(lo, hi, 0,1,2,3,4,5,6,7,8,9,10,11,12,13,14,15);
}

// ---------- projection GEMM: C = A(f32)[M,D] @ W(f32)[D,D] + bias, C in f16 ----------
__global__ __launch_bounds__(256) void proj_f16_kernel(
    const float* __restrict__ A, const float* __restrict__ W,
    const float* __restrict__ bias, h16* __restrict__ C, int transOut)
{
  __shared__ __align__(16) h16 As[128][40];
  __shared__ __align__(16) h16 Bs[128][40];

  const int tid  = threadIdx.x;
  const int lane = tid & 31;
  const int wid  = tid >> 5;
  const int wm   = wid >> 1;
  const int wn   = wid & 1;
  const int mBase = blockIdx.x * 128;
  const int nBase = blockIdx.y * 128;

  v8f acc[2][4];
  const v8f vzero = {0.f,0.f,0.f,0.f,0.f,0.f,0.f,0.f};
  #pragma unroll
  for (int i = 0; i < 2; ++i)
    #pragma unroll
    for (int j = 0; j < 4; ++j) acc[i][j] = vzero;

  for (int k0 = 0; k0 < D; k0 += 32) {
    {  // stage A tile 128x32 (f32 -> f16)
      const int r = tid >> 1;
      const int cseg = (tid & 1) * 16;
      const float* ap = A + (size_t)(mBase + r) * D + k0 + cseg;
      #pragma unroll
      for (int i = 0; i < 4; ++i) {
        float4 f = ((const float4*)ap)[i];
        v4h hh = { (h16)f.x, (h16)f.y, (h16)f.z, (h16)f.w };
        *(v4h*)&As[r][cseg + i*4] = hh;
      }
    }
    {  // stage W tile 32x128 transposed -> Bs[n][k]
      const int kr   = tid >> 3;
      const int nseg = (tid & 7) * 16;
      const float* wp = W + (size_t)(k0 + kr) * D + nBase + nseg;
      #pragma unroll
      for (int i = 0; i < 4; ++i) {
        float4 f = ((const float4*)wp)[i];
        Bs[nseg + i*4 + 0][kr] = (h16)f.x;
        Bs[nseg + i*4 + 1][kr] = (h16)f.y;
        Bs[nseg + i*4 + 2][kr] = (h16)f.z;
        Bs[nseg + i*4 + 3][kr] = (h16)f.w;
      }
    }
    __syncthreads();

    v16h a0 = load_frag_a(&As[0][0], 40, wm*32 +  0, 0, lane);
    v16h a1 = load_frag_a(&As[0][0], 40, wm*32 + 16, 0, lane);
    #pragma unroll
    for (int nt = 0; nt < 4; ++nt) {
      v16h bf = load_frag_b(&Bs[0][0], 40, wn*64 + nt*16, 0, lane);
      acc[0][nt] = wmma_f16(a0, bf, acc[0][nt]);
      acc[1][nt] = wmma_f16(a1, bf, acc[1][nt]);
    }
    __syncthreads();
  }

  const int rhalf = (lane >> 4) << 3;
  if (!transOut) {
    #pragma unroll
    for (int nt = 0; nt < 4; ++nt) {
      const int col = nBase + wn*64 + nt*16 + (lane & 15);
      const float bv = bias[col];
      #pragma unroll
      for (int mt = 0; mt < 2; ++mt)
        #pragma unroll
        for (int g = 0; g < 8; ++g) {
          const int row = mBase + wm*32 + mt*16 + g + rhalf;
          C[(size_t)row * D + col] = (h16)(acc[mt][nt][g] + bv);
        }
    }
  } else {
    #pragma unroll
    for (int nt = 0; nt < 4; ++nt) {
      const int col = nBase + wn*64 + nt*16 + (lane & 15);
      const float bv = bias[col];
      #pragma unroll
      for (int mt = 0; mt < 2; ++mt)
        #pragma unroll
        for (int g = 0; g < 8; ++g) {
          const int row = mBase + wm*32 + mt*16 + g + rhalf;
          const int bb = row >> 11;
          const int zz = row & (T - 1);
          C[((size_t)bb << 21) + ((size_t)col << 11) + zz] = (h16)(acc[mt][nt][g] + bv);
        }
    }
  }
}

// ---------- flash attention: 32 Q-rows per block, 8 waves split Dout ----------
// Q,K: [b][t][d] f16 ; Vt: [b][out][z] f16 ; mask: [T][T] u8 ; Y: [b][t][out] f32
#define DCH 64                         // d super-chunk (halves)
#define KP  72                         // chunk pitch in halves (144B, bank-spread)

__global__ __launch_bounds__(256) void flash_attn_kernel(
    const h16* __restrict__ Q, const h16* __restrict__ K,
    const h16* __restrict__ Vt, const unsigned char* __restrict__ mask,
    float* __restrict__ Y)
{
  __shared__ __align__(16) h16 Ks[2][128][KP];  // double-buffered K chunk
  __shared__ __align__(16) h16 Qs[2][32][KP];   // double-buffered Q chunk
  __shared__ __align__(16) h16 Ps[32][136];     // probs tile
  __shared__ float redmax[32][8];
  __shared__ float redsum[32][8];
  __shared__ float sm_m[32], sm_s[32], sm_new[32], sm_c[32];

  const int tid   = threadIdx.x;
  const int lane  = tid & 31;
  const int wid   = tid >> 5;
  const int rbase = blockIdx.x * 32;
  const int batch = blockIdx.y;
  const size_t boff = (size_t)batch * T * D;
  const h16* Qb = Q  + boff;
  const h16* Kb = K  + boff;
  const h16* Vb = Vt + boff;                    // [out][z], pitch T

  if (tid < 32) { sm_m[tid] = -1e30f; sm_s[tid] = 0.f; }

  v8f o[2][8];
  const v8f vzero = {0.f,0.f,0.f,0.f,0.f,0.f,0.f,0.f};
  #pragma unroll
  for (int i = 0; i < 2; ++i)
    #pragma unroll
    for (int j = 0; j < 8; ++j) o[i][j] = vzero;

  const float C0 = 1.4426950408889634f / 32.0f;  // log2(e)/sqrt(Datt)
  const int rhalf = (lane >> 4) << 3;
  const int nTiles = rbase / 128 + 1;            // causal: skip tiles right of diagonal

  // staging coordinates: per phase each thread issues 5 async B128 copies
  const int krow = tid >> 1, kseg = (tid & 1) * 32;   // K: 2 thr/row x 4 xfers
  const int qrow = tid >> 3, qseg = (tid & 7) * 8;    // Q: 8 thr/row x 1 xfer

  for (int t = 0; t < nTiles; ++t) {
    const int cbase = t * 128;

    // ---- S = Q K^T over D/DCH phases, double-buffered async LDS staging ----
    v8f s0 = vzero, s1 = vzero;
    {  // prologue: stage phase 0 into buffer 0
      const h16* gk = Kb + (size_t)(cbase + krow) * D + kseg;
      unsigned lk = (unsigned)(size_t)&Ks[0][krow][kseg];
      #pragma unroll
      for (int i = 0; i < 4; ++i) ASYNC_G2L_B128(lk + i*16, gk + i*8);
      const h16* gq = Qb + (size_t)(rbase + qrow) * D + qseg;
      ASYNC_G2L_B128((unsigned)(size_t)&Qs[0][qrow][qseg], gq);
    }
    const int P = D / DCH;  // 16
    for (int p = 0; p < P; ++p) {
      const int cur = p & 1;
      if (p + 1 < P) {  // stage next phase into the other buffer
        const int d1 = (p + 1) * DCH;
        const h16* gk = Kb + (size_t)(cbase + krow) * D + d1 + kseg;
        unsigned lk = (unsigned)(size_t)&Ks[cur ^ 1][krow][kseg];
        #pragma unroll
        for (int i = 0; i < 4; ++i) ASYNC_G2L_B128(lk + i*16, gk + i*8);
        const h16* gq = Qb + (size_t)(rbase + qrow) * D + d1 + qseg;
        ASYNC_G2L_B128((unsigned)(size_t)&Qs[cur ^ 1][qrow][qseg], gq);
        WAIT_ASYNC5();   // phase p's copies retired; phase p+1 still in flight
      } else {
        WAIT_ASYNC0();
      }
      __syncthreads();   // all waves' phase-p data visible

      #pragma unroll
      for (int kc = 0; kc < DCH; kc += 32) {
        v16h a0 = load_frag_a(&Qs[cur][0][0], KP,  0, kc, lane);
        v16h a1 = load_frag_a(&Qs[cur][0][0], KP, 16, kc, lane);
        v16h bk = load_frag_b(&Ks[cur][0][0], KP, wid*16, kc, lane);
        s0 = wmma_f16(a0, bk, s0);
        s1 = wmma_f16(a1, bk, s1);
      }
      __syncthreads();   // buffer `cur` may be reissued next iteration
    }
    if (t + 1 < nTiles) {  // warm L2 for next tile's V slice (global_prefetch_b8)
      __builtin_prefetch(Vb + (size_t)(wid*128 + (lane & 15)) * T + cbase + 128, 0, 1);
    }

    // ---- logits + mask ----
    float lf[2][8];
    const int cl = cbase + wid*16 + (lane & 15);
    const bool full = (cbase + 127) <= rbase;
    #pragma unroll
    for (int mt = 0; mt < 2; ++mt) {
      #pragma unroll
      for (int g = 0; g < 8; ++g) {
        float l = (mt ? s1[g] : s0[g]) * C0;
        if (!full) {
          const int row = rbase + mt*16 + g + rhalf;
          if (!mask[(size_t)row * T + cl]) l = -1e30f;
        }
        lf[mt][g] = l;
      }
    }

    // ---- per-row max within wave's 16 cols -> per-wave partials ----
    #pragma unroll
    for (int mt = 0; mt < 2; ++mt) {
      #pragma unroll
      for (int g = 0; g < 8; ++g) {
        float m = lf[mt][g];
        m = fmaxf(m, __shfl_xor(m, 1));
        m = fmaxf(m, __shfl_xor(m, 2));
        m = fmaxf(m, __shfl_xor(m, 4));
        m = fmaxf(m, __shfl_xor(m, 8));
        if ((lane & 15) == 0) redmax[mt*16 + rhalf + g][wid] = m;
      }
    }
    __syncthreads();
    if (tid < 32) {
      float mx = sm_m[tid];
      #pragma unroll
      for (int w = 0; w < 8; ++w) mx = fmaxf(mx, redmax[tid][w]);
      sm_c[tid]   = exp2f(sm_m[tid] - mx);
      sm_new[tid] = mx;
      sm_m[tid]   = mx;
    }
    __syncthreads();

    // ---- P = exp2(l - m) -> LDS, partial row sums ----
    #pragma unroll
    for (int mt = 0; mt < 2; ++mt) {
      #pragma unroll
      for (int g = 0; g < 8; ++g) {
        const int rl = mt*16 + rhalf + g;
        float p = exp2f(lf[mt][g] - sm_new[rl]);
        Ps[rl][wid*16 + (lane & 15)] = (h16)p;
        float s = p;
        s += __shfl_xor(s, 1);
        s += __shfl_xor(s, 2);
        s += __shfl_xor(s, 4);
        s += __shfl_xor(s, 8);
        if ((lane & 15) == 0) redsum[rl][wid] = s;
      }
    }
    __syncthreads();
    if (tid < 32) {
      float s = sm_s[tid] * sm_c[tid];
      #pragma unroll
      for (int w = 0; w < 8; ++w) s += redsum[tid][w];
      sm_s[tid] = s;
    }

    // ---- rescale O ----
    #pragma unroll
    for (int mt = 0; mt < 2; ++mt)
      #pragma unroll
      for (int g = 0; g < 8; ++g) {
        const float c = sm_c[mt*16 + rhalf + g];
        #pragma unroll
        for (int nt = 0; nt < 8; ++nt) o[mt][nt][g] *= c;
      }

    // ---- O += P * V  (A = P from LDS, B = V^T from L2-resident global) ----
    #pragma unroll
    for (int zc = 0; zc < 4; ++zc) {
      v16h pa0 = load_frag_a(&Ps[0][0], 136,  0, zc*32, lane);
      v16h pa1 = load_frag_a(&Ps[0][0], 136, 16, zc*32, lane);
      #pragma unroll
      for (int nt = 0; nt < 8; ++nt) {
        v16h vb = load_frag_b(Vb, T, wid*128 + nt*16, cbase + zc*32, lane);
        o[0][nt] = wmma_f16(pa0, vb, o[0][nt]);
        o[1][nt] = wmma_f16(pa1, vb, o[1][nt]);
      }
    }
    __syncthreads();   // protect Ps/red arrays before next tile
  }

  // ---- epilogue: normalize and store f32 ----
  #pragma unroll
  for (int mt = 0; mt < 2; ++mt) {
    #pragma unroll
    for (int g = 0; g < 8; ++g) {
      const int rl  = mt*16 + rhalf + g;
      const float inv = 1.0f / sm_s[rl];
      const int row = rbase + rl;
      #pragma unroll
      for (int nt = 0; nt < 8; ++nt) {
        const int col = wid*128 + nt*16 + (lane & 15);
        Y[((size_t)batch * T + row) * D + col] = o[mt][nt][g] * inv;
      }
    }
  }
}

// ---------- host launch ----------
extern "C" void kernel_launch(void* const* d_in, const int* in_sizes, int n_in,
                              void* d_out, int out_size, void* d_ws, size_t ws_size,
                              hipStream_t stream) {
  (void)in_sizes; (void)n_in; (void)out_size; (void)ws_size;
  const float* x  = (const float*)d_in[0];
  const float* z  = (const float*)d_in[1];
  const float* Wq = (const float*)d_in[2];
  const float* bq = (const float*)d_in[3];
  const float* Wk = (const float*)d_in[4];
  const float* bk = (const float*)d_in[5];
  const float* Wv = (const float*)d_in[6];
  const float* bv = (const float*)d_in[7];
  const unsigned char* mask = (const unsigned char*)d_in[8];
  float* y = (float*)d_out;

  const size_t elems = (size_t)NB * T * D;
  h16* Qh = (h16*)d_ws;
  h16* Kh = Qh + elems;
  h16* Vh = Kh + elems;                      // transposed [b][out][z]

  dim3 gProj((NB * T) / 128, D / 128);
  proj_f16_kernel<<<gProj, 256, 0, stream>>>(x, Wq, bq, Qh, 0);
  proj_f16_kernel<<<gProj, 256, 0, stream>>>(z, Wk, bk, Kh, 0);
  proj_f16_kernel<<<gProj, 256, 0, stream>>>(z, Wv, bv, Vh, 1);

  dim3 gF(T / 32, NB);
  flash_attn_kernel<<<gF, 256, 0, stream>>>(Qh, Kh, Vh, mask, y);
}
